// LatticeConvolution_57114475102427
// MI455X (gfx1250) — matrix operand order repaired and checked by reading, not compiled, not run on hardware
//
#include <hip/hip_runtime.h>
#include <hip/hip_bf16.h>

// ---------------------------------------------------------------------------
// LatticeConvolution on gfx1250 (MI455X): implicit GEMM via v_wmma_f32_16x16x32_bf16
// with Tensor-Data-Mover (TDM) double-buffered staging of activation tiles.
//   M = COUT(192), N = output pixels (per-row tiles of 64), K = CIN(192) x 7 taps
//   fp32 inputs DMA'd to LDS (f32), converted to bf16 compute layout in LDS,
//   f32 accumulation, f32 output.
// ---------------------------------------------------------------------------

#define B_    8
#define CIN_  192
#define COUT_ 192
#define H_    128
#define W_    128
#define OH_   127
#define OW_   127

#define CI_CHUNK   32                   // K per WMMA
#define N_CHUNKS   (CIN_ / CI_CHUNK)    // 6
#define TILE_W     64                   // pixels per block (N tile)
#define TILE_CO    64                   // COUT per block (M tile)
#define PADC       66                   // staged input columns (TILE_W + 2 halo)
#define ROWS0      3                    // x0 halo rows
#define ROWS1      2                    // x1 halo rows

typedef __attribute__((ext_vector_type(16))) __bf16 v16bf;
typedef __attribute__((ext_vector_type(8)))  float  v8f;

// TDM descriptor groups (clang-23 / amdgpu-toolchain 6-arg builtin form:
//   (uint32x4 g0, int32x8 g1, int32x4 g2, int32x4 g3, int32x8 g4, i32 cpol))
typedef __attribute__((ext_vector_type(4))) unsigned int tdm_v4u;
typedef __attribute__((ext_vector_type(8))) int          tdm_v8i;
typedef __attribute__((ext_vector_type(4))) int          tdm_v4i;

union Frag {
    v16bf bf;
    uint4 u4[2];
};

__device__ __forceinline__ unsigned short f2bf(float x) {
    unsigned int u = __float_as_uint(x);
    unsigned int r = (u + 0x7FFFu + ((u >> 16) & 1u)) >> 16;   // RNE
    return (unsigned short)r;
}

// Generic pointer to an LDS object: low 32 bits are the LDS byte offset
// (flat LDS aperture keeps the offset in ADDR[31:0]).
#define LDS_BYTE_OFFSET(p) ((unsigned)(unsigned long long)(uintptr_t)(p))

// Issue a TDM 3-D tile load: tile_cols (contiguous) x tile_rows (stride
// row_stride elems) x tile_chans (stride chan_stride elems), f32 elements,
// linear fill into LDS at lds_off. tensor_dim* = elements remaining from the
// tile start so hardware zero-fills out-of-bounds reads.
__device__ __forceinline__ void tdm_load_tile_f32(const float* gptr, unsigned lds_off,
                                                  unsigned cols_rem, unsigned rows_rem,
                                                  unsigned chans_rem,
                                                  unsigned tile_cols, unsigned tile_rows,
                                                  unsigned tile_chans,
                                                  unsigned long long row_stride,
                                                  unsigned long long chan_stride) {
    const unsigned long long ga = (unsigned long long)(uintptr_t)gptr;

    tdm_v4u g0;
    g0.x = 1u;                                            // count=1 (valid), user mode
    g0.y = lds_off;                                       // lds_addr (bytes)
    g0.z = (unsigned)(ga & 0xFFFFFFFFull);                // global_addr[31:0]
    g0.w = (unsigned)((ga >> 32) & 0x1FFFFFFull)          // global_addr[56:32]
         | (2u << 30);                                    // type = 2 ("image")

    tdm_v8i g1;
    g1[0] = (int)(2u << 16);                              // data_size = 4 bytes
    g1[1] = (int)((cols_rem & 0xFFFFu) << 16);            // tensor_dim0[15:0]
    g1[2] = (int)((cols_rem >> 16)                        // tensor_dim0[31:16]
                | ((rows_rem & 0xFFFFu) << 16));          // tensor_dim1[15:0]
    g1[3] = (int)(((rows_rem >> 16) & 0xFFFFu)            // tensor_dim1[31:16]
                | (tile_cols << 16));                     // tile_dim0
    g1[4] = (int)((tile_rows & 0xFFFFu)                   // tile_dim1
                | (tile_chans << 16));                    // tile_dim2
    g1[5] = (int)(row_stride & 0xFFFFFFFFull);            // tensor_dim0_stride[31:0]
    g1[6] = (int)(((row_stride >> 32) & 0xFFFFull)        // tensor_dim0_stride[47:32]
                | ((chan_stride & 0xFFFFull) << 16));     // tensor_dim1_stride[15:0]
    g1[7] = (int)((chan_stride >> 16) & 0xFFFFFFFFull);   // tensor_dim1_stride[47:16]

    tdm_v4i g2;
    g2[0] = (int)chans_rem;                               // tensor_dim2
    g2[1] = 0;                                            // tensor_dim3 / no iterate
    g2[2] = 0;                                            // tensor_dim2_stride lo (unused, tile_dim3=0)
    g2[3] = 0;                                            // tile_dim3 = 0 (unused)

    tdm_v4i g3 = {0, 0, 0, 0};
    tdm_v8i g4 = {0, 0, 0, 0, 0, 0, 0, 0};

    __builtin_amdgcn_tensor_load_to_lds(g0, g1, g2, g3, g4, 0);
}

// B fragment: 32(K=ci) x 16(N=pixel). Lane L: n = L&15, K-half = L>>4.
// LDS layout: [row][col][ci] (ci innermost, 32 contiguous bf16 per (row,col)).
__device__ __forceinline__ v16bf load_b_frag(const unsigned short* rowbase, int c, int lane) {
    const int n  = lane & 15;
    const int kb = (lane >> 4) << 4;           // 0 or 16
    const unsigned short* p = rowbase + (c + n) * CI_CHUNK + kb;
    Frag f;
    f.u4[0] = *(const uint4*)(p);
    f.u4[1] = *(const uint4*)(p + 8);
    return f.bf;
}

// A fragment: 16(M=co) x 32(K=ci). Lane L: m = L&15.
//   lanes 0-15 hold K = 0..15, lanes 16-31 hold K = 16..31.
// LDS layout: [tap][co][ci] (ci innermost).
__device__ __forceinline__ v16bf load_a_frag(const unsigned short* tapbase, int co_local, int lane) {
    const int m  = lane & 15;
    const int hi = (lane >> 4) << 4;           // 0 or 16
    const unsigned short* p = tapbase + (co_local + m) * CI_CHUNK + hi;
    Frag f;
    f.u4[0] = *(const uint4*)(p);
    f.u4[1] = *(const uint4*)(p + 8);
    return f.bf;
}

#define WMMA_BF16(A, Bm, C) \
    __builtin_amdgcn_wmma_f32_16x16x32_bf16(false, (A), false, (Bm), (short)0, (C), false, false)

__global__ void __launch_bounds__(256)
LatticeConvolution_57114475102427_kernel(const float* __restrict__ x0,
                                         const float* __restrict__ x1,
                                         const float* __restrict__ w0,
                                         const float* __restrict__ w1,
                                         const float* __restrict__ bias,
                                         float* __restrict__ out) {
    // f32 staging buffers filled by TDM (layout: [ci][row][col], col innermost)
    __shared__ __align__(16) float stage_x0[2][CI_CHUNK * ROWS0 * PADC];
    __shared__ __align__(16) float stage_x1[2][CI_CHUNK * ROWS1 * PADC];
    // bf16 compute buffers (ci innermost)
    __shared__ __align__(16) unsigned short lds_x0[ROWS0 * PADC * CI_CHUNK];
    __shared__ __align__(16) unsigned short lds_x1[ROWS1 * PADC * CI_CHUNK];
    __shared__ __align__(16) unsigned short lds_w0[4 * TILE_CO * CI_CHUNK];  // K0: 4 taps
    __shared__ __align__(16) unsigned short lds_w1[3 * TILE_CO * CI_CHUNK];  // K1: 3 taps

    const int tid  = threadIdx.x;
    const int lane = tid & 31;
    const int wid  = tid >> 5;
    const int wave_m = wid >> 2;      // 0..1 -> 32 co
    const int wave_n = wid & 3;       // 0..3 -> 16 pixels

    const int bid    = blockIdx.x;
    const int co_blk = bid % 3;
    const int w_blk  = (bid / 3) & 1;
    const int h      = (bid / 6) % OH_;
    const int b      = bid / (6 * OH_);

    const int co0   = co_blk * TILE_CO;
    const int wbase = w_blk * TILE_W;
    const int cb    = wave_n * 16;    // this wave's column base inside the staged tile

    const unsigned cols_rem = (unsigned)(W_ - wbase);
    const unsigned rows_rem = (unsigned)(H_ - h);

    v8f acc0[2] = {{0.f}, {0.f}};     // out0, M-subtiles 0/1
    v8f acc1[2] = {{0.f}, {0.f}};     // out1

    // ---- prime the pipeline: TDM chunk 0 into buffer 0 (wave 0 only; TDM
    //      ignores EXEC so it must be branch-gated at wave granularity)
    if (wid == 0) {
        const float* g0p = &x0[(((size_t)b * CIN_ + 0) * H_ + h) * W_ + wbase];
        const float* g1p = &x1[(((size_t)b * CIN_ + 0) * H_ + h) * W_ + wbase];
        tdm_load_tile_f32(g0p, LDS_BYTE_OFFSET(&stage_x0[0][0]),
                          cols_rem, rows_rem, (unsigned)CIN_,
                          PADC, ROWS0, CI_CHUNK, W_, (unsigned long long)H_ * W_);
        tdm_load_tile_f32(g1p, LDS_BYTE_OFFSET(&stage_x1[0][0]),
                          cols_rem, rows_rem, (unsigned)CIN_,
                          PADC, ROWS1, CI_CHUNK, W_, (unsigned long long)H_ * W_);
        __builtin_amdgcn_s_wait_tensorcnt(0);
    }
    __syncthreads();

    for (int kc = 0; kc < N_CHUNKS; ++kc) {
        const int ci0 = kc * CI_CHUNK;
        const int cur = kc & 1;

        // ---- convert TDM f32 stage ([ci][row][col]) -> bf16 compute layout ([row][col][ci])
        for (int idx = tid; idx < CI_CHUNK * ROWS0 * PADC; idx += 256) {
            const int c  = idx % PADC;
            const int r  = (idx / PADC) % ROWS0;
            const int ci = idx / (PADC * ROWS0);
            lds_x0[(r * PADC + c) * CI_CHUNK + ci] = f2bf(stage_x0[cur][idx]);
        }
        for (int idx = tid; idx < CI_CHUNK * ROWS1 * PADC; idx += 256) {
            const int c  = idx % PADC;
            const int r  = (idx / PADC) % ROWS1;
            const int ci = idx / (PADC * ROWS1);
            lds_x1[(r * PADC + c) * CI_CHUNK + ci] = f2bf(stage_x1[cur][idx]);
        }
        // ---- stage w0 (COUT, CIN, 4) -> [tap][co][ci]  (L2-resident, direct path)
        for (int idx = tid; idx < 4 * TILE_CO * CI_CHUNK; idx += 256) {
            const int ci = idx & (CI_CHUNK - 1);
            const int t2 = idx >> 5;
            const int co = t2 & (TILE_CO - 1);
            const int t  = t2 >> 6;
            const float v = w0[((size_t)(co0 + co) * CIN_ + (ci0 + ci)) * 4 + t];
            lds_w0[idx] = f2bf(v);
        }
        // ---- stage w1 (COUT, CIN, 3) -> [tap][co][ci]
        for (int idx = tid; idx < 3 * TILE_CO * CI_CHUNK; idx += 256) {
            const int ci = idx & (CI_CHUNK - 1);
            const int t2 = idx >> 5;
            const int co = t2 & (TILE_CO - 1);
            const int t  = t2 >> 6;
            const float v = w1[((size_t)(co0 + co) * CIN_ + (ci0 + ci)) * 3 + t];
            lds_w1[idx] = f2bf(v);
        }
        __syncthreads();

        // ---- kick off next chunk's TDM into the other staging buffer
        if (kc + 1 < N_CHUNKS && wid == 0) {
            const int cin = ci0 + CI_CHUNK;
            const float* g0p = &x0[(((size_t)b * CIN_ + cin) * H_ + h) * W_ + wbase];
            const float* g1p = &x1[(((size_t)b * CIN_ + cin) * H_ + h) * W_ + wbase];
            tdm_load_tile_f32(g0p, LDS_BYTE_OFFSET(&stage_x0[cur ^ 1][0]),
                              cols_rem, rows_rem, (unsigned)(CIN_ - cin),
                              PADC, ROWS0, CI_CHUNK, W_, (unsigned long long)H_ * W_);
            tdm_load_tile_f32(g1p, LDS_BYTE_OFFSET(&stage_x1[cur ^ 1][0]),
                              cols_rem, rows_rem, (unsigned)(CIN_ - cin),
                              PADC, ROWS1, CI_CHUNK, W_, (unsigned long long)H_ * W_);
        }
        // Prefetch next chunk's weights into L2 while WMMAs run.
        if (kc + 1 < N_CHUNKS) {
            __builtin_prefetch(&w0[((size_t)(co0 + (lane & 31)) * CIN_ + ci0 + CI_CHUNK) * 4], 0, 1);
            __builtin_prefetch(&w1[((size_t)(co0 + (lane & 31)) * CIN_ + ci0 + CI_CHUNK) * 3], 0, 1);
        }

        // ---- B fragments (shared across both outputs): 10 distinct taps
        const v16bf b0_00 = load_b_frag(lds_x0 + 0 * PADC * CI_CHUNK, cb + 0, lane);
        const v16bf b0_01 = load_b_frag(lds_x0 + 0 * PADC * CI_CHUNK, cb + 1, lane);
        const v16bf b0_10 = load_b_frag(lds_x0 + 1 * PADC * CI_CHUNK, cb + 0, lane);
        const v16bf b0_11 = load_b_frag(lds_x0 + 1 * PADC * CI_CHUNK, cb + 1, lane);
        const v16bf b0_12 = load_b_frag(lds_x0 + 1 * PADC * CI_CHUNK, cb + 2, lane);
        const v16bf b0_21 = load_b_frag(lds_x0 + 2 * PADC * CI_CHUNK, cb + 1, lane);
        const v16bf b1_00 = load_b_frag(lds_x1 + 0 * PADC * CI_CHUNK, cb + 0, lane);
        const v16bf b1_01 = load_b_frag(lds_x1 + 0 * PADC * CI_CHUNK, cb + 1, lane);
        const v16bf b1_10 = load_b_frag(lds_x1 + 1 * PADC * CI_CHUNK, cb + 0, lane);
        const v16bf b1_11 = load_b_frag(lds_x1 + 1 * PADC * CI_CHUNK, cb + 1, lane);

#pragma unroll
        for (int ms = 0; ms < 2; ++ms) {
            const int cl = wave_m * 32 + ms * 16;   // co_local base
            const v16bf a0_00 = load_a_frag(lds_w0 + 0 * TILE_CO * CI_CHUNK, cl, lane);
            const v16bf a0_01 = load_a_frag(lds_w0 + 1 * TILE_CO * CI_CHUNK, cl, lane);
            const v16bf a0_10 = load_a_frag(lds_w0 + 2 * TILE_CO * CI_CHUNK, cl, lane);
            const v16bf a0_11 = load_a_frag(lds_w0 + 3 * TILE_CO * CI_CHUNK, cl, lane);
            const v16bf a1_0  = load_a_frag(lds_w1 + 0 * TILE_CO * CI_CHUNK, cl, lane);
            const v16bf a1_1  = load_a_frag(lds_w1 + 1 * TILE_CO * CI_CHUNK, cl, lane);
            const v16bf a1_2  = load_a_frag(lds_w1 + 2 * TILE_CO * CI_CHUNK, cl, lane);

            // out0 = K0 * x0 + K1 * x1
            acc0[ms] = WMMA_BF16(a0_00, b0_00, acc0[ms]);
            acc0[ms] = WMMA_BF16(a0_01, b0_01, acc0[ms]);
            acc0[ms] = WMMA_BF16(a0_10, b0_10, acc0[ms]);
            acc0[ms] = WMMA_BF16(a0_11, b0_11, acc0[ms]);
            acc0[ms] = WMMA_BF16(a1_0,  b1_00, acc0[ms]);
            acc0[ms] = WMMA_BF16(a1_1,  b1_01, acc0[ms]);
            acc0[ms] = WMMA_BF16(a1_2,  b1_10, acc0[ms]);

            // out1 = K0 * x1 + K1 * x0_shift(+1,+1)
            acc1[ms] = WMMA_BF16(a0_00, b1_00, acc1[ms]);
            acc1[ms] = WMMA_BF16(a0_01, b1_01, acc1[ms]);
            acc1[ms] = WMMA_BF16(a0_10, b1_10, acc1[ms]);
            acc1[ms] = WMMA_BF16(a0_11, b1_11, acc1[ms]);
            acc1[ms] = WMMA_BF16(a1_0,  b0_11, acc1[ms]);
            acc1[ms] = WMMA_BF16(a1_1,  b0_12, acc1[ms]);
            acc1[ms] = WMMA_BF16(a1_2,  b0_21, acc1[ms]);
        }

        // ---- wait for the in-flight TDM before the next iteration consumes it
        if (kc + 1 < N_CHUNKS && wid == 0) {
            __builtin_amdgcn_s_wait_tensorcnt(0);
        }
        __syncthreads();
    }

    // ---- epilogue: C layout (VGPR r, lanes 0-15: M=r; lanes 16-31: M=r+8), add bias
    const int n_w  = lane & 15;
    const int m_hi = (lane >> 4) << 3;   // 0 or 8
    const int gw   = wbase + wave_n * 16 + n_w;
    if (gw < OW_) {
#pragma unroll
        for (int ms = 0; ms < 2; ++ms) {
            const int cobase = co0 + wave_m * 32 + ms * 16;
#pragma unroll
            for (int r = 0; r < 8; ++r) {
                const int co = cobase + r + m_hi;
                const float bv = bias[co];
                const size_t o0 = ((((size_t)0 * B_ + b) * COUT_ + co) * OH_ + h) * OW_ + gw;
                const size_t o1 = ((((size_t)1 * B_ + b) * COUT_ + co) * OH_ + h) * OW_ + gw;
                out[o0] = acc0[ms][r] + bv;
                out[o1] = acc1[ms][r] + bv;
            }
        }
    }
}

extern "C" void kernel_launch(void* const* d_in, const int* in_sizes, int n_in,
                              void* d_out, int out_size, void* d_ws, size_t ws_size,
                              hipStream_t stream) {
    (void)in_sizes; (void)n_in; (void)out_size; (void)d_ws; (void)ws_size;
    const float* x0   = (const float*)d_in[0];
    const float* x1   = (const float*)d_in[1];
    const float* w0   = (const float*)d_in[2];
    const float* w1   = (const float*)d_in[3];
    const float* bias = (const float*)d_in[4];
    float* out = (float*)d_out;

    // grid = co_blocks(3) * w_blocks(2) * OH(127) * B(8) = 6096 workgroups
    dim3 grid(3 * 2 * OH_ * B_);
    dim3 block(256);
    LatticeConvolution_57114475102427_kernel<<<grid, block, 0, stream>>>(x0, x1, w0, w1, bias, out);
}
